// LearnablePQ_73813307949412
// MI455X (gfx1250) — matrix operand order repaired and checked by reading, not compiled, not run on hardware
//
#include <hip/hip_runtime.h>
#include <hip/hip_bf16.h>

// Problem constants (reference: B=32768, DIM=768, M=8, KS=256, SUBDIM=96)
static constexpr int BN   = 32768;
static constexpr int DIMN = 768;
static constexpr int MN   = 8;
static constexpr int KSN  = 256;
static constexpr int SUBN = 96;
#define EPSF 1e-12f

typedef __attribute__((ext_vector_type(8)))  __bf16 v8bf;
typedef __attribute__((ext_vector_type(16))) __bf16 v16bf;
typedef __attribute__((ext_vector_type(8)))  float  v8f;

// ---------------------------------------------------------------------------
// Kernel 0: zero the usage accumulator region of d_out (harness poisons it)
// ---------------------------------------------------------------------------
__global__ void k_zero_usage(float* __restrict__ u) {
  int i = blockIdx.x * blockDim.x + threadIdx.x;
  if (i < MN * KSN) u[i] = 0.0f;
}

// ---------------------------------------------------------------------------
// Kernel 1: row-normalize x and convert to bf16.  z = x / max(||x||, eps)
// One block (256 threads) per row of 768 floats.
// ---------------------------------------------------------------------------
__global__ __launch_bounds__(256) void k_norm_x(const float* __restrict__ x,
                                                __bf16* __restrict__ zh) {
  __shared__ float red[256];
  const int row = blockIdx.x;
  const int tid = threadIdx.x;
  const float* xr = x + (size_t)row * DIMN;
  float v0 = xr[tid], v1 = xr[tid + 256], v2 = xr[tid + 512];
  red[tid] = v0 * v0 + v1 * v1 + v2 * v2;
  __syncthreads();
  for (int o = 128; o > 0; o >>= 1) {
    if (tid < o) red[tid] += red[tid + o];
    __syncthreads();
  }
  const float s = 1.0f / fmaxf(sqrtf(red[0]), EPSF);
  __bf16* zr = zh + (size_t)row * DIMN;
  zr[tid]       = (__bf16)(v0 * s);
  zr[tid + 256] = (__bf16)(v1 * s);
  zr[tid + 512] = (__bf16)(v2 * s);
}

// ---------------------------------------------------------------------------
// Kernel 2: convert W (row-major fp32) to bf16 row-major (B-operand for GEMM1)
// ---------------------------------------------------------------------------
__global__ void k_conv_w(const float* __restrict__ W, __bf16* __restrict__ Wh) {
  int i = blockIdx.x * blockDim.x + threadIdx.x;
  if (i < DIMN * DIMN) Wh[i] = (__bf16)W[i];
}

// ---------------------------------------------------------------------------
// Kernel 3: normalize codebooks; emit fp32 copy (for gather), bf16 copy (for
// WMMA) and cc = ||c||^2.  One wave32 per (m,k) row of 96 elems.
// ---------------------------------------------------------------------------
__global__ __launch_bounds__(256) void k_prep_cb(const float* __restrict__ cb,
                                                 float*  __restrict__ cbn,
                                                 __bf16* __restrict__ cbh,
                                                 float*  __restrict__ cc) {
  const int w = threadIdx.x >> 5, lane = threadIdx.x & 31;
  const int row = blockIdx.x * 8 + w;               // 0 .. M*KS-1
  const float* c = cb + (size_t)row * SUBN;
  float a0 = c[lane], a1 = c[lane + 32], a2 = c[lane + 64];
  float ss = a0 * a0 + a1 * a1 + a2 * a2;
  for (int o = 16; o > 0; o >>= 1) ss += __shfl_xor(ss, o, 32);
  const float inv = 1.0f / fmaxf(sqrtf(ss), EPSF);
  const float n0 = a0 * inv, n1 = a1 * inv, n2 = a2 * inv;
  float* pn = cbn + (size_t)row * SUBN;
  __bf16* ph = cbh + (size_t)row * SUBN;
  pn[lane] = n0; pn[lane + 32] = n1; pn[lane + 64] = n2;
  ph[lane] = (__bf16)n0; ph[lane + 32] = (__bf16)n1; ph[lane + 64] = (__bf16)n2;
  float c2 = n0 * n0 + n1 * n1 + n2 * n2;
  for (int o = 16; o > 0; o >>= 1) c2 += __shfl_xor(c2, o, 32);
  if (lane == 0) cc[row] = c2;
}

// ---------------------------------------------------------------------------
// Kernel 4: z2 = z @ W^T  via v_wmma_f32_16x16x32_bf16, register-double-
// buffered so step-(k+1) fragment loads overlap step-k matrix ops.
// Block = 256 thr = 8 waves laid out 4(M) x 2(N): block tile 64 x 128.
// Wave tile 16 x 64 (4 accumulators).  K loop: 768/32 = 24 steps.
// ---------------------------------------------------------------------------
__global__ __launch_bounds__(256) void k_gemm1(const __bf16* __restrict__ zh,
                                               const __bf16* __restrict__ Wh,
                                               __bf16* __restrict__ z2h) {
  const int tid = threadIdx.x;
  const int w = tid >> 5, lane = tid & 31;
  const int lmod = lane & 15, lh = lane >> 4;
  const int waveM = w & 3, waveN = w >> 2;
  const int row0 = blockIdx.x * 64 + waveM * 16;
  const int col0 = blockIdx.y * 128 + waveN * 64;

  v8f acc0 = {}, acc1 = {}, acc2 = {}, acc3 = {};
  const __bf16* arow = zh + (size_t)(row0 + lmod) * DIMN;
  const __bf16* b0p = Wh + (size_t)(col0 +  0 + lmod) * DIMN + 16 * lh;
  const __bf16* b1p = Wh + (size_t)(col0 + 16 + lmod) * DIMN + 16 * lh;
  const __bf16* b2p = Wh + (size_t)(col0 + 32 + lmod) * DIMN + 16 * lh;
  const __bf16* b3p = Wh + (size_t)(col0 + 48 + lmod) * DIMN + 16 * lh;

  union AV { v16bf v; v8bf h[2]; };
  AV a_c, a_n;
  v16bf bc0, bc1, bc2, bc3, bn0, bn1, bn2, bn3;

  // Prologue: load k-step 0 fragments.
  a_c.h[0] = *(const v8bf*)(arow + 16 * lh);
  a_c.h[1] = *(const v8bf*)(arow + 8 + 16 * lh);
  bc0 = *(const v16bf*)(b0p);
  bc1 = *(const v16bf*)(b1p);
  bc2 = *(const v16bf*)(b2p);
  bc3 = *(const v16bf*)(b3p);

#pragma unroll
  for (int kb = 0; kb < DIMN; kb += 32) {
    const int kn = kb + 32;
    if (kn < DIMN) {
      // Issue next-step loads BEFORE this step's matrix ops.
      __builtin_prefetch((const void*)(arow + kn + 32), 0, 3);
      a_n.h[0] = *(const v8bf*)(arow + kn + 16 * lh);
      a_n.h[1] = *(const v8bf*)(arow + kn + 8 + 16 * lh);
      bn0 = *(const v16bf*)(b0p + kn);
      bn1 = *(const v16bf*)(b1p + kn);
      bn2 = *(const v16bf*)(b2p + kn);
      bn3 = *(const v16bf*)(b3p + kn);
    }
    acc0 = __builtin_amdgcn_wmma_f32_16x16x32_bf16(false, a_c.v, false, bc0, (short)0, acc0, false, false);
    acc1 = __builtin_amdgcn_wmma_f32_16x16x32_bf16(false, a_c.v, false, bc1, (short)0, acc1, false, false);
    acc2 = __builtin_amdgcn_wmma_f32_16x16x32_bf16(false, a_c.v, false, bc2, (short)0, acc2, false, false);
    acc3 = __builtin_amdgcn_wmma_f32_16x16x32_bf16(false, a_c.v, false, bc3, (short)0, acc3, false, false);
    a_c = a_n;
    bc0 = bn0; bc1 = bn1; bc2 = bn2; bc3 = bn3;
  }
  // C/D layout: lane l -> n = l%16; vgpr r -> m = r + 8*(l/16)
#pragma unroll
  for (int r = 0; r < 8; r++) {
    __bf16* zr = z2h + (size_t)(row0 + r + 8 * lh) * DIMN;
    zr[col0 +  0 + lmod] = (__bf16)acc0[r];
    zr[col0 + 16 + lmod] = (__bf16)acc1[r];
    zr[col0 + 32 + lmod] = (__bf16)acc2[r];
    zr[col0 + 48 + lmod] = (__bf16)acc3[r];
  }
}

// ---------------------------------------------------------------------------
// Kernel 5: fused  zc = z2[:,m] @ cb[m]^T  (WMMA, K=96)  ->  gumbel softmax
// (shift-invariant: the ||z||^2 term cancels in softmax/argmax and is dropped)
// -> argmax -> straight-through prob -> prob/codes/usage/quantized outputs.
// Block handles 32 rows x 256 codes for one subspace m.
// Waves: 2(M) x 4(N), wave tile 16 x 64, 3 k-steps of 32 (K=96),
// register-double-buffered like k_gemm1.
// ---------------------------------------------------------------------------
__global__ __launch_bounds__(256) void k_gemm2_fused(
    const __bf16* __restrict__ z2h, const __bf16* __restrict__ cbh,
    const float* __restrict__ cbn, const float* __restrict__ cc,
    const float* __restrict__ gu,
    float* __restrict__ out_quant, float* __restrict__ out_codes,
    float* __restrict__ out_prob, float* __restrict__ out_usage) {
  __shared__ float zc[32][KSN];     // 32 KB
  __shared__ float s_usage[KSN];    // 1 KB

  const int tid = threadIdx.x;
  const int w = tid >> 5, lane = tid & 31;
  const int lmod = lane & 15, lh = lane >> 4;
  const int m = blockIdx.y;
  const int brow0 = blockIdx.x * 32;
  const int waveM = w >> 2, waveN = w & 3;
  const int row0 = brow0 + waveM * 16;
  const int col0 = waveN * 64;

  if (tid < KSN) s_usage[tid] = 0.0f;

  v8f acc0 = {}, acc1 = {}, acc2 = {}, acc3 = {};
  const __bf16* arow = z2h + (size_t)(row0 + lmod) * DIMN + m * SUBN;
  const __bf16* bb = cbh + (size_t)m * KSN * SUBN + 16 * lh;
  const __bf16* b0p = bb + (size_t)(col0 +  0 + lmod) * SUBN;
  const __bf16* b1p = bb + (size_t)(col0 + 16 + lmod) * SUBN;
  const __bf16* b2p = bb + (size_t)(col0 + 32 + lmod) * SUBN;
  const __bf16* b3p = bb + (size_t)(col0 + 48 + lmod) * SUBN;

  union AV { v16bf v; v8bf h[2]; };
  AV a_c, a_n;
  v16bf bc0, bc1, bc2, bc3, bn0, bn1, bn2, bn3;
  a_c.h[0] = *(const v8bf*)(arow + 16 * lh);
  a_c.h[1] = *(const v8bf*)(arow + 8 + 16 * lh);
  bc0 = *(const v16bf*)(b0p);
  bc1 = *(const v16bf*)(b1p);
  bc2 = *(const v16bf*)(b2p);
  bc3 = *(const v16bf*)(b3p);

#pragma unroll
  for (int kb = 0; kb < SUBN; kb += 32) {
    const int kn = kb + 32;
    if (kn < SUBN) {
      a_n.h[0] = *(const v8bf*)(arow + kn + 16 * lh);
      a_n.h[1] = *(const v8bf*)(arow + kn + 8 + 16 * lh);
      bn0 = *(const v16bf*)(b0p + kn);
      bn1 = *(const v16bf*)(b1p + kn);
      bn2 = *(const v16bf*)(b2p + kn);
      bn3 = *(const v16bf*)(b3p + kn);
    }
    acc0 = __builtin_amdgcn_wmma_f32_16x16x32_bf16(false, a_c.v, false, bc0, (short)0, acc0, false, false);
    acc1 = __builtin_amdgcn_wmma_f32_16x16x32_bf16(false, a_c.v, false, bc1, (short)0, acc1, false, false);
    acc2 = __builtin_amdgcn_wmma_f32_16x16x32_bf16(false, a_c.v, false, bc2, (short)0, acc2, false, false);
    acc3 = __builtin_amdgcn_wmma_f32_16x16x32_bf16(false, a_c.v, false, bc3, (short)0, acc3, false, false);
    a_c = a_n;
    bc0 = bn0; bc1 = bn1; bc2 = bn2; bc3 = bn3;
  }
#pragma unroll
  for (int r = 0; r < 8; r++) {
    const int lr = waveM * 16 + r + 8 * lh;
    zc[lr][col0 +  0 + lmod] = acc0[r];
    zc[lr][col0 + 16 + lmod] = acc1[r];
    zc[lr][col0 + 32 + lmod] = acc2[r];
    zc[lr][col0 + 48 + lmod] = acc3[r];
  }
  __syncthreads();

  // Phase 3: wave w owns rows w*4 .. w*4+3 of the 32-row tile.
  for (int rr = 0; rr < 4; rr++) {
    const int lr = w * 4 + rr;
    const int b = brow0 + lr;
    const float* gur = gu + ((size_t)b * MN + m) * KSN;
    float tv[8];
    float best = -3.0e38f;
    int bidx = 0;
#pragma unroll
    for (int i = 0; i < 8; i++) {
      const int k = lane + 32 * i;
      const float u = gur[k];
      const float g = -__logf(-__logf(u));
      const float t = 2.0f * zc[lr][k] - cc[m * KSN + k] + g;   // + const(b,m)
      tv[i] = t;
      if (t > best) { best = t; bidx = k; }
    }
    for (int o = 16; o > 0; o >>= 1) {
      const float ov = __shfl_xor(best, o, 32);
      const int oi = __shfl_xor(bidx, o, 32);
      if (ov > best || (ov == best && oi < bidx)) { best = ov; bidx = oi; }
    }
    float ev[8];
    float s = 0.0f;
#pragma unroll
    for (int i = 0; i < 8; i++) { ev[i] = __expf(tv[i] - best); s += ev[i]; }
    for (int o = 16; o > 0; o >>= 1) s += __shfl_xor(s, o, 32);
    const float invs = 1.0f / s;

    float* pr = out_prob + ((size_t)b * MN + m) * KSN;
#pragma unroll
    for (int i = 0; i < 8; i++) {
      const int k = lane + 32 * i;
      const float y = ev[i] * invs;
      const float h = (k == bidx) ? 1.0f : 0.0f;
      const float p = (h - y) + y;        // straight-through forward value
      pr[k] = p;
      atomicAdd(&s_usage[k], p);          // ds_add_f32
    }
    if (lane == 0) out_codes[(size_t)b * MN + m] = (float)bidx;
    // quantized = selected normalized codeword (prob is one-hot forward)
    const float* q = cbn + ((size_t)m * KSN + bidx) * SUBN;
    float* oq = out_quant + (size_t)b * DIMN + m * SUBN;
    oq[lane] = q[lane]; oq[lane + 32] = q[lane + 32]; oq[lane + 64] = q[lane + 64];
  }
  __syncthreads();
  if (tid < KSN)
    atomicAdd(&out_usage[m * KSN + tid], s_usage[tid] * (1.0f / (float)BN));
}

// ---------------------------------------------------------------------------
extern "C" void kernel_launch(void* const* d_in, const int* in_sizes, int n_in,
                              void* d_out, int out_size, void* d_ws, size_t ws_size,
                              hipStream_t stream) {
  (void)in_sizes; (void)n_in; (void)out_size; (void)ws_size;
  const float* x  = (const float*)d_in[0];
  const float* W  = (const float*)d_in[1];
  const float* cb = (const float*)d_in[2];
  const float* gu = (const float*)d_in[3];

  float* out = (float*)d_out;
  float* out_quant = out;                                          // B*DIM
  float* out_codes = out_quant + (size_t)BN * DIMN;                // B*M
  float* out_prob  = out_codes + (size_t)BN * MN;                  // B*M*KS
  float* out_usage = out_prob  + (size_t)BN * MN * KSN;            // M*KS

  char* ws = (char*)d_ws;                         // offsets 256B aligned
  __bf16* zh  = (__bf16*)(ws);                    // 50,331,648 B
  __bf16* z2h = (__bf16*)(ws + 50331648);         // 50,331,648 B
  __bf16* Wh  = (__bf16*)(ws + 100663296);        //  1,179,648 B
  __bf16* cbh = (__bf16*)(ws + 101842944);        //    393,216 B
  float*  cbn = (float*) (ws + 102236160);        //    786,432 B
  float*  cc  = (float*) (ws + 103022592);        //      8,192 B

  hipLaunchKernelGGL(k_zero_usage, dim3(8), dim3(256), 0, stream, out_usage);
  hipLaunchKernelGGL(k_norm_x, dim3(BN), dim3(256), 0, stream, x, zh);
  hipLaunchKernelGGL(k_conv_w, dim3((DIMN * DIMN + 255) / 256), dim3(256), 0, stream, W, Wh);
  hipLaunchKernelGGL(k_prep_cb, dim3((MN * KSN) / 8), dim3(256), 0, stream, cb, cbn, cbh, cc);
  hipLaunchKernelGGL(k_gemm1, dim3(BN / 64, DIMN / 128), dim3(256), 0, stream, zh, Wh, z2h);
  hipLaunchKernelGGL(k_gemm2_fused, dim3(BN / 32, MN), dim3(256), 0, stream,
                     z2h, cbh, cbn, cc, gu, out_quant, out_codes, out_prob, out_usage);
}
